// CausalTransformer_46866683134049
// MI455X (gfx1250) — matrix operand order repaired and checked by reading, compiled, not run on hardware
//
#include <hip/hip_runtime.h>
#include <hip/hip_bf16.h>
#include <cstdint>

// ---------------------------------------------------------------------------
// CDNA5 / gfx1250 bf16 WMMA transformer forward
//   L=4, D=1024, H=16, HD=64, F=4096, B=2, T=2048  -> NROWS = B*T = 4096
// ---------------------------------------------------------------------------

#define DMODEL 1024
#define NHEAD  16
#define HDIM   64
#define FDIM   4096
#define SEQ    2048
#define NROWS  4096   // B*T

typedef __attribute__((ext_vector_type(16))) __bf16 v16bf;
typedef __attribute__((ext_vector_type(8)))  float  v8f;

union FragU {                 // one WMMA A/B operand: 16 bf16 = 32 bytes
    v16bf v;
    uint4 q[2];
};

__device__ __forceinline__ unsigned short f32_to_bf16(float f) {
    union { float f; unsigned u; } a; a.f = f;
    unsigned u = a.u;
    u += 0x7fffu + ((u >> 16) & 1u);   // round-to-nearest-even
    return (unsigned short)(u >> 16);
}

#define WMMA_BF16(A, B, C) \
    __builtin_amdgcn_wmma_f32_16x16x32_bf16(false, (A), false, (B), (short)0, (C), false, false)

// ---- CDNA5 async global->LDS path (ASYNCcnt), guarded; falls back to a
// ---- register-prefetch software pipeline if the builtin is absent.
#if defined(__has_builtin)
# if __has_builtin(__builtin_amdgcn_global_load_async_to_lds_b128)
#  define USE_ASYNC_LDS 1
# else
#  define USE_ASYNC_LDS 0
# endif
#else
# define USE_ASYNC_LDS 0
#endif

#if USE_ASYNC_LDS
typedef int v4i_ext __attribute__((ext_vector_type(4)));
typedef __attribute__((address_space(1))) v4i_ext as1_v4i;   // global
typedef __attribute__((address_space(3))) v4i_ext as3_v4i;   // LDS
__device__ __forceinline__ void async_b128(const unsigned short* g, unsigned short* l) {
    // global: flat address == global address numerically (cast via integer);
    // LDS: proper addrspacecast generic -> as(3).
    __builtin_amdgcn_global_load_async_to_lds_b128(
        (as1_v4i*)(uintptr_t)g, (as3_v4i*)l, 0, 0);
}
#endif

__device__ __forceinline__ void wait_async_zero() {
#if USE_ASYNC_LDS
# if __has_builtin(__builtin_amdgcn_s_wait_asynccnt)
    __builtin_amdgcn_s_wait_asynccnt(0);
# else
    asm volatile("s_wait_asynccnt 0x0" ::: "memory");
# endif
#endif
}

// ---------------------------------------------------------------------------
// f32 -> bf16 weight conversion (per layer, once)
// ---------------------------------------------------------------------------
__global__ __launch_bounds__(256) void xfrm_cvt_kernel(
    const float* __restrict__ src, unsigned short* __restrict__ dst, int n4)
{
    const int i = blockIdx.x * 256 + threadIdx.x;
    if (i < n4) {
        const float4 f = ((const float4*)src)[i];
        uint2 pk;
        pk.x = (unsigned)f32_to_bf16(f.x) | ((unsigned)f32_to_bf16(f.y) << 16);
        pk.y = (unsigned)f32_to_bf16(f.z) | ((unsigned)f32_to_bf16(f.w) << 16);
        ((uint2*)dst)[i] = pk;
    }
}

// ---------------------------------------------------------------------------
// LayerNorm: rows of 1024 f32 -> bf16 (obf) or f32 (of32)
// ---------------------------------------------------------------------------
__global__ __launch_bounds__(256) void xfrm_ln_kernel(
    const float* __restrict__ x, const float* __restrict__ g,
    const float* __restrict__ b, unsigned short* __restrict__ obf,
    float* __restrict__ of32)
{
    const int row = blockIdx.x;
    const int tid = threadIdx.x;
    const float* xr = x + (size_t)row * DMODEL;

    float v[4];
    float s = 0.f;
#pragma unroll
    for (int i = 0; i < 4; ++i) { v[i] = xr[tid + i * 256]; s += v[i]; }

    __shared__ float r1[8], r2[8];
#pragma unroll
    for (int o = 16; o > 0; o >>= 1) s += __shfl_xor(s, o, 32);
    if ((tid & 31) == 0) r1[tid >> 5] = s;
    __syncthreads();
    float mean = 0.f;
#pragma unroll
    for (int k = 0; k < 8; ++k) mean += r1[k];
    mean *= (1.f / (float)DMODEL);

    float qacc = 0.f;
#pragma unroll
    for (int i = 0; i < 4; ++i) { float d = v[i] - mean; qacc += d * d; }
#pragma unroll
    for (int o = 16; o > 0; o >>= 1) qacc += __shfl_xor(qacc, o, 32);
    if ((tid & 31) == 0) r2[tid >> 5] = qacc;
    __syncthreads();
    float var = 0.f;
#pragma unroll
    for (int k = 0; k < 8; ++k) var += r2[k];
    var *= (1.f / (float)DMODEL);
    const float rstd = rsqrtf(var + 1e-5f);

#pragma unroll
    for (int i = 0; i < 4; ++i) {
        const int c = tid + i * 256;
        const float y = (v[i] - mean) * rstd * g[c] + b[c];
        if (obf)  obf[(size_t)row * DMODEL + c] = f32_to_bf16(y);
        else      of32[(size_t)row * DMODEL + c] = y;
    }
}

// ---------------------------------------------------------------------------
// GEMM: out[n,m] = sum_k A[n,k]*W[m,k] + bias[m]  (A, W bf16; f32 accumulate)
//   mode 0: store bf16 | mode 1: exact GELU -> bf16 | mode 2: +resid -> f32
// Block: 512 threads (16 waves), tile 256(N) x 128(M), K-step 32.
// Wave grid 4x4, wave tile 64x32 (4x2 WMMA accumulators).
// Double-buffered LDS, software-pipelined staging (async DMA if available).
// ---------------------------------------------------------------------------
__global__ __launch_bounds__(512) void xfrm_gemm_kernel(
    const unsigned short* __restrict__ A,   // NROWS x K (bf16)
    const unsigned short* __restrict__ W,   // M x K (bf16)
    const float* __restrict__ bias,         // M
    const float* __restrict__ resid,        // NROWS x M (mode 2)
    void* __restrict__ out,
    int M, int K, int mode)
{
    __shared__ __align__(16) unsigned short sA[2][256 * 40];
    __shared__ __align__(16) unsigned short sW[2][128 * 40];

    const int tid  = threadIdx.x;
    const int lane = tid & 31;
    const int w    = tid >> 5;
    const int hl   = lane >> 4;
    const int li   = lane & 15;

    const int n0 = blockIdx.x * 256;
    const int m0 = blockIdx.y * 128;
    const int wr = (w >> 2) * 64;   // wave row offset (0,64,128,192)
    const int wc = (w & 3) * 32;    // wave col offset (0,32,64,96)

    // staging geometry: A = 1024 b128 chunks (2/thread), W = 512 chunks (1/thread)
    const int ar  = tid >> 2;            // 0..127 (and +128)
    const int as  = (tid & 3) * 8;
    const unsigned short* Ag0 = A + (size_t)(n0 + ar) * K + as;
    const unsigned short* Ag1 = A + (size_t)(n0 + ar + 128) * K + as;
    const unsigned short* Wg  = W + (size_t)(m0 + ar) * K + as;   // ar<128 rows
    const int adst0 = ar * 40 + as;
    const int adst1 = (ar + 128) * 40 + as;
    const int wdst  = ar * 40 + as;

#if !USE_ASYNC_LDS
    uint4 pfA0, pfA1, pfW;
#endif

    v8f acc[4][2];
#pragma unroll
    for (int a = 0; a < 4; ++a)
#pragma unroll
        for (int c = 0; c < 2; ++c)
#pragma unroll
            for (int j = 0; j < 8; ++j) acc[a][c][j] = 0.f;

    const int nk = K >> 5;

    auto issue = [&](int k, int buf) {
        const int k0 = k * 32;
#if USE_ASYNC_LDS
        async_b128(Ag0 + k0, &sA[buf][adst0]);
        async_b128(Ag1 + k0, &sA[buf][adst1]);
        async_b128(Wg + k0, &sW[buf][wdst]);
#else
        pfA0 = *(const uint4*)(Ag0 + k0);
        pfA1 = *(const uint4*)(Ag1 + k0);
        pfW  = *(const uint4*)(Wg + k0);
#endif
    };
    auto commit = [&](int buf) {
#if !USE_ASYNC_LDS
        *(uint4*)(&sA[buf][adst0]) = pfA0;
        *(uint4*)(&sA[buf][adst1]) = pfA1;
        *(uint4*)(&sW[buf][wdst])  = pfW;
#endif
        (void)buf;
    };

    issue(0, 0);
    for (int k = 0; k < nk; ++k) {
        const int cur = k & 1;
        commit(cur);
        wait_async_zero();
        __syncthreads();
        if (k + 1 < nk) issue(k + 1, cur ^ 1);

        FragU af[4], bf2[2];
#pragma unroll
        for (int a = 0; a < 4; ++a) {
            const unsigned short* base = &sA[cur][(wr + a * 16 + li) * 40];
            af[a].q[0] = *(const uint4*)(base + hl * 16);
            af[a].q[1] = *(const uint4*)(base + hl * 16 + 8);
        }
#pragma unroll
        for (int c = 0; c < 2; ++c) {
            const unsigned short* base = &sW[cur][(wc + c * 16 + li) * 40 + hl * 16];
            bf2[c].q[0] = *(const uint4*)(base);
            bf2[c].q[1] = *(const uint4*)(base + 8);
        }
#pragma unroll
        for (int a = 0; a < 4; ++a)
#pragma unroll
            for (int c = 0; c < 2; ++c)
                acc[a][c] = WMMA_BF16(af[a].v, bf2[c].v, acc[a][c]);
    }

    // ---- epilogue
#pragma unroll
    for (int c = 0; c < 2; ++c) {
        const int col = m0 + wc + c * 16 + li;
        const float bv = bias[col];
#pragma unroll
        for (int a = 0; a < 4; ++a) {
#pragma unroll
            for (int j = 0; j < 8; ++j) {
                const int rowi = n0 + wr + a * 16 + j + 8 * hl;
                float val = acc[a][c][j] + bv;
                const size_t idx = (size_t)rowi * M + col;
                if (mode == 0) {
                    ((unsigned short*)out)[idx] = f32_to_bf16(val);
                } else if (mode == 1) {
                    val = 0.5f * val * (1.0f + erff(val * 0.70710678118654752f));
                    ((unsigned short*)out)[idx] = f32_to_bf16(val);
                } else {
                    ((float*)out)[idx] = val + resid[idx];
                }
            }
        }
    }
}

// ---------------------------------------------------------------------------
// Flash attention (causal + ALiBi), block = (64-query tile, b*H+h), 4 waves.
// Double-buffered K/V staging pipelined against WMMA + softmax.
// qkv: (B*T) x 3072 bf16 ; attnout: (B*T) x 1024 bf16
// ---------------------------------------------------------------------------
__global__ __launch_bounds__(128) void xfrm_attn_kernel(
    const unsigned short* __restrict__ qkv,
    unsigned short* __restrict__ attnout)
{
    const int qt  = blockIdx.x;
    const int bh  = blockIdx.y;
    const int bb  = bh >> 4;
    const int h   = bh & 15;
    const int tid = threadIdx.x;
    const int lane = tid & 31;
    const int w    = tid >> 5;
    const int hl   = lane >> 4;
    const int li   = lane & 15;

    const float slope = exp2f(-0.5f * (float)(h + 1));   // H=16 ALiBi slopes
    const size_t rowbase = (size_t)bb * SEQ;
    const int q0 = qt * 64;

    __shared__ __align__(16) unsigned short sQ[64 * 72];
    __shared__ __align__(16) unsigned short sK[2][64 * 72];
    __shared__ __align__(16) unsigned short sVt[2][64 * 72];
    __shared__ __align__(16) unsigned short sP[4][16 * 72];

    // ---- stage Q tile (64 x 64 bf16), visible after first barrier
#pragma unroll
    for (int it = 0; it < 4; ++it) {
        const int task = tid + it * 128;
        const int r = task >> 3, c = (task & 7) * 8;
        *(uint4*)(sQ + r * 72 + c) =
            *(const uint4*)(qkv + (rowbase + q0 + r) * 3072 + h * 64 + c);
    }

    // staging geometry
    const int kr = tid >> 3, kc = (tid & 7) * 8;      // K: 4 chunks (rows +16*it)
    const int vr = tid >> 4, vc = (tid & 15) * 4;     // V: 8 chunks (rows +8*it)
#if !USE_ASYNC_LDS
    uint4 kreg[4];
#endif
    uint2 varr[8];

    auto issueKV = [&](int kt, int buf) {
        const size_t krow = rowbase + (size_t)kt * 64;
#pragma unroll
        for (int it = 0; it < 4; ++it) {
            const int r = kr + it * 16;
#if USE_ASYNC_LDS
            async_b128(qkv + (krow + r) * 3072 + 1024 + h * 64 + kc,
                       &sK[buf][r * 72 + kc]);
#else
            kreg[it] = *(const uint4*)(qkv + (krow + r) * 3072 + 1024 + h * 64 + kc);
#endif
        }
#pragma unroll
        for (int it = 0; it < 8; ++it) {
            const int r = vr + it * 8;
            varr[it] = *(const uint2*)(qkv + (krow + r) * 3072 + 2048 + h * 64 + vc);
        }
        (void)buf;
    };
    auto commitKV = [&](int buf) {
#if !USE_ASYNC_LDS
#pragma unroll
        for (int it = 0; it < 4; ++it)
            *(uint4*)(&sK[buf][(kr + it * 16) * 72 + kc]) = kreg[it];
#endif
#pragma unroll
        for (int it = 0; it < 8; ++it) {       // transposed V: sVt[d][n]
            const int r = vr + it * 8;
            const unsigned x0 = varr[it].x, x1 = varr[it].y;
            sVt[buf][(vc + 0) * 72 + r] = (unsigned short)(x0 & 0xffff);
            sVt[buf][(vc + 1) * 72 + r] = (unsigned short)(x0 >> 16);
            sVt[buf][(vc + 2) * 72 + r] = (unsigned short)(x1 & 0xffff);
            sVt[buf][(vc + 3) * 72 + r] = (unsigned short)(x1 >> 16);
        }
    };

    float m_run[8], l_run[8];
    v8f o[4];
#pragma unroll
    for (int j = 0; j < 8; ++j) { m_run[j] = -__builtin_inff(); l_run[j] = 0.f; }
#pragma unroll
    for (int d = 0; d < 4; ++d)
#pragma unroll
        for (int j = 0; j < 8; ++j) o[d][j] = 0.f;

    const int ktiles = qt + 1;
    issueKV(0, 0);
    for (int kt = 0; kt < ktiles; ++kt) {
        const int cur = kt & 1;
        commitKV(cur);
        wait_async_zero();
        __syncthreads();
        if (kt + 1 < ktiles) issueKV(kt + 1, cur ^ 1);

        // ---- S = Q K^T : wave owns 16 query rows, 64 key cols
        v8f s[4];
#pragma unroll
        for (int c = 0; c < 4; ++c)
#pragma unroll
            for (int j = 0; j < 8; ++j) s[c][j] = 0.f;

#pragma unroll
        for (int ks = 0; ks < 2; ++ks) {
            FragU aq;
            const unsigned short* ab = sQ + (w * 16 + li) * 72 + ks * 32;
            aq.q[0] = *(const uint4*)(ab + hl * 16);
            aq.q[1] = *(const uint4*)(ab + hl * 16 + 8);
#pragma unroll
            for (int c = 0; c < 4; ++c) {
                FragU bk;
                const unsigned short* kb = &sK[cur][(c * 16 + li) * 72 + ks * 32 + hl * 16];
                bk.q[0] = *(const uint4*)(kb);
                bk.q[1] = *(const uint4*)(kb + 8);
                s[c] = WMMA_BF16(aq.v, bk.v, s[c]);
            }
        }

        // ---- scale + ALiBi + causal mask, online softmax
        const int qrow0 = q0 + w * 16 + 8 * hl;
        const int kcol0 = kt * 64 + li;
        float sv[4][8];
        float rmax[8];
#pragma unroll
        for (int j = 0; j < 8; ++j) rmax[j] = -__builtin_inff();
#pragma unroll
        for (int c = 0; c < 4; ++c) {
#pragma unroll
            for (int j = 0; j < 8; ++j) {
                const int qr = qrow0 + j;
                const int kc2 = kcol0 + c * 16;
                float val = s[c][j] * 0.125f + slope * (float)(kc2 - qr);
                if (kc2 > qr) val = -__builtin_inff();
                sv[c][j] = val;
                rmax[j] = fmaxf(rmax[j], val);
            }
        }
#pragma unroll
        for (int j = 0; j < 8; ++j)
#pragma unroll
            for (int off = 1; off < 16; off <<= 1)
                rmax[j] = fmaxf(rmax[j], __shfl_xor(rmax[j], off, 16));

        float alpha[8];
#pragma unroll
        for (int j = 0; j < 8; ++j) {
            const float mn = fmaxf(m_run[j], rmax[j]);
            alpha[j] = __expf(m_run[j] - mn);
            m_run[j] = mn;
        }
        float rsum[8];
#pragma unroll
        for (int j = 0; j < 8; ++j) rsum[j] = 0.f;
#pragma unroll
        for (int c = 0; c < 4; ++c)
#pragma unroll
            for (int j = 0; j < 8; ++j) {
                const float p = __expf(sv[c][j] - m_run[j]);
                sv[c][j] = p;
                rsum[j] += p;
            }
#pragma unroll
        for (int j = 0; j < 8; ++j)
#pragma unroll
            for (int off = 1; off < 16; off <<= 1)
                rsum[j] += __shfl_xor(rsum[j], off, 16);
#pragma unroll
        for (int j = 0; j < 8; ++j) l_run[j] = l_run[j] * alpha[j] + rsum[j];

#pragma unroll
        for (int d = 0; d < 4; ++d)
#pragma unroll
            for (int j = 0; j < 8; ++j) o[d][j] *= alpha[j];

        // ---- P -> LDS (bf16), then O += P @ V
        unsigned short* pbuf = sP[w];
#pragma unroll
        for (int c = 0; c < 4; ++c)
#pragma unroll
            for (int j = 0; j < 8; ++j)
                pbuf[(j + 8 * hl) * 72 + c * 16 + li] = f32_to_bf16(sv[c][j]);

#pragma unroll
        for (int ns = 0; ns < 2; ++ns) {
            FragU ap;
            const unsigned short* ab = pbuf + li * 72 + ns * 32;
            ap.q[0] = *(const uint4*)(ab + hl * 16);
            ap.q[1] = *(const uint4*)(ab + hl * 16 + 8);
#pragma unroll
            for (int d = 0; d < 4; ++d) {
                FragU bv;
                const unsigned short* vb = &sVt[cur][(d * 16 + li) * 72 + ns * 32 + hl * 16];
                bv.q[0] = *(const uint4*)(vb);
                bv.q[1] = *(const uint4*)(vb + 8);
                o[d] = WMMA_BF16(ap.v, bv.v, o[d]);
            }
        }
    }

    // ---- epilogue: O / l -> bf16
#pragma unroll
    for (int d = 0; d < 4; ++d)
#pragma unroll
        for (int j = 0; j < 8; ++j) {
            const float val = o[d][j] / l_run[j];
            const int r = q0 + w * 16 + j + 8 * hl;
            attnout[(rowbase + r) * DMODEL + h * 64 + d * 16 + li] = f32_to_bf16(val);
        }
}

// ---------------------------------------------------------------------------
// Host-side orchestration
// ---------------------------------------------------------------------------
extern "C" void kernel_launch(void* const* d_in, const int* in_sizes, int n_in,
                              void* d_out, int out_size, void* d_ws, size_t ws_size,
                              hipStream_t stream) {
    const float* x     = (const float*)d_in[0];
    const float* in_w  = (const float*)d_in[1];
    const float* in_b  = (const float*)d_in[2];
    const float* out_w = (const float*)d_in[3];
    const float* out_b = (const float*)d_in[4];
    const float* f1w   = (const float*)d_in[5];
    const float* f1b   = (const float*)d_in[6];
    const float* f2w   = (const float*)d_in[7];
    const float* f2b   = (const float*)d_in[8];
    const float* ln1w  = (const float*)d_in[9];
    const float* ln1b  = (const float*)d_in[10];
    const float* ln2w  = (const float*)d_in[11];
    const float* ln2b  = (const float*)d_in[12];
    const float* finw  = (const float*)d_in[13];
    const float* finb  = (const float*)d_in[14];

    char* ws = (char*)d_ws;
    size_t off = 0;
    float* xres = (float*)(ws + off);                   off += (size_t)NROWS * DMODEL * 4;
    unsigned short* xn   = (unsigned short*)(ws + off); off += (size_t)NROWS * DMODEL * 2;
    unsigned short* qkv  = (unsigned short*)(ws + off); off += (size_t)NROWS * 3 * DMODEL * 2;
    unsigned short* attn = (unsigned short*)(ws + off); off += (size_t)NROWS * DMODEL * 2;
    unsigned short* hbuf = (unsigned short*)(ws + off); off += (size_t)NROWS * FDIM * 2;
    unsigned short* wqkv = (unsigned short*)(ws + off); off += (size_t)3 * DMODEL * DMODEL * 2;
    unsigned short* wo   = (unsigned short*)(ws + off); off += (size_t)DMODEL * DMODEL * 2;
    unsigned short* w1   = (unsigned short*)(ws + off); off += (size_t)FDIM * DMODEL * 2;
    unsigned short* w2   = (unsigned short*)(ws + off); off += (size_t)DMODEL * FDIM * 2;

    (void)hipMemcpyAsync(xres, x, (size_t)NROWS * DMODEL * sizeof(float),
                         hipMemcpyDeviceToDevice, stream);

    const int n4_qkv = (3 * DMODEL * DMODEL) / 4;
    const int n4_o   = (DMODEL * DMODEL) / 4;
    const int n4_f   = (FDIM * DMODEL) / 4;

    for (int i = 0; i < 4; ++i) {
        // one-shot f32 -> bf16 weight conversion for this layer
        xfrm_cvt_kernel<<<(n4_qkv + 255) / 256, 256, 0, stream>>>(
            in_w + (size_t)i * 3 * DMODEL * DMODEL, wqkv, n4_qkv);
        xfrm_cvt_kernel<<<(n4_o + 255) / 256, 256, 0, stream>>>(
            out_w + (size_t)i * DMODEL * DMODEL, wo, n4_o);
        xfrm_cvt_kernel<<<(n4_f + 255) / 256, 256, 0, stream>>>(
            f1w + (size_t)i * FDIM * DMODEL, w1, n4_f);
        xfrm_cvt_kernel<<<(n4_f + 255) / 256, 256, 0, stream>>>(
            f2w + (size_t)i * DMODEL * FDIM, w2, n4_f);

        xfrm_ln_kernel<<<NROWS, 256, 0, stream>>>(
            xres, ln1w + i * DMODEL, ln1b + i * DMODEL, xn, nullptr);
        xfrm_gemm_kernel<<<dim3(NROWS / 256, (3 * DMODEL) / 128), 512, 0, stream>>>(
            xn, wqkv, in_b + (size_t)i * 3 * DMODEL, nullptr, qkv, 3 * DMODEL, DMODEL, 0);
        xfrm_attn_kernel<<<dim3(SEQ / 64, 2 * NHEAD), 128, 0, stream>>>(qkv, attn);
        xfrm_gemm_kernel<<<dim3(NROWS / 256, DMODEL / 128), 512, 0, stream>>>(
            attn, wo, out_b + (size_t)i * DMODEL, xres, xres, DMODEL, DMODEL, 2);
        xfrm_ln_kernel<<<NROWS, 256, 0, stream>>>(
            xres, ln2w + i * DMODEL, ln2b + i * DMODEL, xn, nullptr);
        xfrm_gemm_kernel<<<dim3(NROWS / 256, FDIM / 128), 512, 0, stream>>>(
            xn, w1, f1b + (size_t)i * FDIM, nullptr, hbuf, FDIM, DMODEL, 1);
        xfrm_gemm_kernel<<<dim3(NROWS / 256, DMODEL / 128), 512, 0, stream>>>(
            hbuf, w2, f2b + (size_t)i * DMODEL, xres, xres, DMODEL, FDIM, 2);
    }
    xfrm_ln_kernel<<<NROWS, 256, 0, stream>>>(xres, finw, finb, nullptr, (float*)d_out);
}
